// FrequencySelfAttention3_35124242547048
// MI455X (gfx1250) — matrix operand order repaired and checked by reading, not compiled, not run on hardware
//
#include <hip/hip_runtime.h>
#include <cstdint>

#define B_ 8
#define C_ 128
#define N_ 4096
#define CHUNK 32
#define KP 136   // kbuf/xs row stride in halves (128 + 8 pad)
#define VP 40    // vbuf row stride in halves   (32 + 8 pad)
#define PP 40    // pbuf row stride in halves

typedef _Float16 v16h __attribute__((ext_vector_type(16)));
typedef float    v8f  __attribute__((ext_vector_type(8)));
typedef unsigned int u32x4 __attribute__((ext_vector_type(4)));
typedef int      i32x4 __attribute__((ext_vector_type(4)));
typedef int      i32x8 __attribute__((ext_vector_type(8)));

union HF16 { v16h v; u32x4 q[2]; _Float16 h[16]; };
union HF8  { u32x4 q; _Float16 h[8]; };

__device__ inline v8f wmma_f16(v16h a, v16h b, v8f c) {
  // D = A(16x32 f16) * B(32x16 f16) + C(16x16 f32)
  return __builtin_amdgcn_wmma_f32_16x16x32_f16(false, a, false, b, (short)0, c,
                                                false, false);
}

// A-fragment: two runs of 8 contiguous halves, 8 halves apart.
__device__ inline v16h frag_2x8(const _Float16* p0, const _Float16* p1) {
  HF16 t;
  t.q[0] = *reinterpret_cast<const u32x4*>(p0);
  t.q[1] = *reinterpret_cast<const u32x4*>(p1);
  return t.v;
}
// B-fragment: one run of 16 contiguous halves.
__device__ inline v16h frag_16(const _Float16* p) { return frag_2x8(p, p + 8); }

__device__ inline float rmax16(float x) {
  #pragma unroll
  for (int m = 1; m < 16; m <<= 1) x = fmaxf(x, __shfl_xor(x, m, 32));
  return x;
}
__device__ inline float rsum16(float x) {
  #pragma unroll
  for (int m = 1; m < 16; m <<= 1) x += __shfl_xor(x, m, 32);
  return x;
}

// ---- Tensor Data Mover: build D# (groups per CDNA5 ISA) and
// issue TENSOR_LOAD_TO_LDS. 2D tile, 2-byte elements, optional LDS padding.
// 6-arg builtin form (clang-23 / therock headers).
__device__ inline void tdm_load2d(unsigned lds_addr, const void* g,
                                  unsigned tile0, unsigned tile1,
                                  unsigned tdim0, unsigned tdim1,
                                  unsigned stride0,
                                  unsigned pad_code, unsigned pad_amt_m1) {
  unsigned long long ga = (unsigned long long)(uintptr_t)g;
  u32x4 g0;
  g0[0] = 1u;                       // count = 1 valid descriptor
  g0[1] = lds_addr;                 // lds_addr (bytes)
  g0[2] = (unsigned)ga;             // global_addr[31:0]
  g0[3] = (unsigned)((ga >> 32) & 0x01ffffffu) | 0x80000000u;  // addr[56:32] | type=2
  i32x8 g1;
  g1[0] = (int)((1u << 16)          // data_size = 1 (2 bytes)
              | (1u << 20)          // pad_enable
              | (pad_code << 22)    // pad_interval
              | (pad_amt_m1 << 25));// pad_amount (N-1 DWORDs)
  g1[1] = (int)((tdim0 & 0xffffu) << 16);
  g1[2] = (int)(((tdim0 >> 16) & 0xffffu) | ((tdim1 & 0xffffu) << 16));
  g1[3] = (int)(((tdim1 >> 16) & 0xffffu) | ((tile0 & 0xffffu) << 16));
  g1[4] = (int)(tile1 & 0xffffu);   // tile_dim1, tile_dim2 = 0
  g1[5] = (int)stride0;             // tensor_dim0_stride[31:0]
  g1[6] = 0;
  g1[7] = 0;
  i32x4 gz4 = {0, 0, 0, 0};
  i32x8 gz8 = {0, 0, 0, 0, 0, 0, 0, 0};
  __builtin_amdgcn_tensor_load_to_lds(g0, g1, gz4, gz4, gz8, 0);
}

// ============================================================================
// Kernel 1: refined = fw[c]*x (FFT block collapses exactly to this), then
// Q/K/V 1x1-conv projections via WMMA. Q scaled by 1/sqrt(C).
// Q,K -> token-major [B][N][C] f16 ; V -> channel-major [B][C][N] f16.
// ============================================================================
__global__ __launch_bounds__(256) void proj_qkv(
    const float* __restrict__ x,  const float* __restrict__ fw,
    const float* __restrict__ wq, const float* __restrict__ bq,
    const float* __restrict__ wk, const float* __restrict__ bk,
    const float* __restrict__ wv, const float* __restrict__ bv,
    _Float16* __restrict__ Qh, _Float16* __restrict__ Kh,
    _Float16* __restrict__ Vc) {
  __shared__ alignas(16) _Float16 xs[64 * KP];  // [n][c], padded
  const int tid  = threadIdx.x;
  const int lane = tid & 31, wid = tid >> 5;
  const int l16  = lane & 15, hi = lane >> 4;
  const int b    = blockIdx.x >> 6;
  const int n0   = (blockIdx.x & 63) * 64;

  // Stage X chunk [128c x 64n] -> LDS token-major with fw folded in.
  for (int i = tid; i < C_ * 64; i += 256) {
    int c = i >> 6, n = i & 63;
    xs[n * KP + c] = (_Float16)(x[((size_t)(b * C_ + c)) * N_ + n0 + n] * fw[c]);
  }
  __syncthreads();

  const int   o_base = wid * 16;
  const float qs     = 0.08838834764831845f;  // 1/sqrt(128)

  for (int m = 0; m < 3; ++m) {
    const float* W    = (m == 0) ? wq : (m == 1) ? wk : wv;
    const float* bias = (m == 0) ? bq : (m == 1) ? bk : bv;
    v16h af[4];  // A = W rows [o_base..o_base+16), K = c
    {
      const int o = o_base + l16;
      const int base = hi ? 16 : 0;
      #pragma unroll
      for (int kk = 0; kk < 4; ++kk) {
        HF16 t;
        #pragma unroll
        for (int j = 0; j < 8; ++j) {
          t.h[j]     = (_Float16)W[o * C_ + kk * 32 + base + j];
          t.h[8 + j] = (_Float16)W[o * C_ + kk * 32 + base + 8 + j];
        }
        af[kk] = t.v;
      }
    }
    #pragma unroll
    for (int sub = 0; sub < 4; ++sub) {
      v8f acc = {0.f, 0.f, 0.f, 0.f, 0.f, 0.f, 0.f, 0.f};
      const _Float16* xr = xs + (sub * 16 + l16) * KP + (hi ? 16 : 0);
      #pragma unroll
      for (int kk = 0; kk < 4; ++kk)
        acc = wmma_f16(af[kk], frag_16(xr + kk * 32), acc);
      const int n  = n0 + sub * 16 + l16;
      const int o0 = o_base + 8 * hi;  // D rows r+8*hi
      if (m < 2) {
        HF8 r;
        #pragma unroll
        for (int rr = 0; rr < 8; ++rr) {
          float v = acc[rr] + bias[o0 + rr];
          if (m == 0) v *= qs;
          r.h[rr] = (_Float16)v;
        }
        _Float16* dst = ((m == 0) ? Qh : Kh) + ((size_t)(b * N_ + n)) * C_ + o0;
        *reinterpret_cast<u32x4*>(dst) = r.q;  // 8 contiguous halves
      } else {
        #pragma unroll
        for (int rr = 0; rr < 8; ++rr)
          Vc[((size_t)(b * C_ + o0 + rr)) * N_ + n] = (_Float16)(acc[rr] + bias[o0 + rr]);
      }
    }
  }
}

// ============================================================================
// Kernel 2: flash attention. 8 waves/block, 16 query rows per wave.
// K/V chunks staged via TDM (double-buffered, LDS-padded for bank-conflict-
// free WMMA fragment reads). Online softmax across 4096 keys.
// ============================================================================
__global__ __launch_bounds__(256) void flash_attn(
    const _Float16* __restrict__ Qh, const _Float16* __restrict__ Kh,
    const _Float16* __restrict__ Vc, _Float16* __restrict__ Oh) {
  __shared__ alignas(16) _Float16 kbuf[2][CHUNK * KP];  // [m][c] padded
  __shared__ alignas(16) _Float16 vbuf[2][C_ * VP];     // [c][m] padded
  __shared__ alignas(16) _Float16 pbuf[8][16 * PP];     // per-wave P transpose

  const int tid   = threadIdx.x;
  const int lane  = tid & 31, wid = tid >> 5;
  const int l16   = lane & 15, hi = lane >> 4;
  const int b     = blockIdx.x >> 5;
  const int nrow0 = (blockIdx.x & 31) * 128 + wid * 16;

  // Q A-fragments (rows = this wave's 16 tokens, K-dim = c in 4 chunks of 32).
  v16h qf[4];
  {
    const _Float16* qrow = Qh + ((size_t)(b * N_ + nrow0 + l16)) * C_ + (hi ? 16 : 0);
    #pragma unroll
    for (int kk = 0; kk < 4; ++kk)
      qf[kk] = frag_2x8(qrow + kk * 32, qrow + kk * 32 + 8);
  }

  const v8f vz = {0.f, 0.f, 0.f, 0.f, 0.f, 0.f, 0.f, 0.f};
  v8f o_acc[8];
  #pragma unroll
  for (int i = 0; i < 8; ++i) o_acc[i] = vz;
  float m_i[8], l_i[8];
  #pragma unroll
  for (int r = 0; r < 8; ++r) { m_i[r] = -1e30f; l_i[r] = 0.f; }

  auto issue = [&](int chunk) {
    const int bufi = chunk & 1;
    // K chunk: contiguous [32 tokens][128 c], pad 4 DWORDs every 64 DWORDs
    // (one 256B token row) -> LDS row stride 136 halves.
    tdm_load2d((unsigned)(uintptr_t)&kbuf[bufi][0],
               Kh + ((size_t)(b * N_ + chunk * CHUNK)) * C_,
               /*tile0*/ CHUNK * C_, /*tile1*/ 1,
               /*tdim0*/ CHUNK * C_, /*tdim1*/ 1,
               /*stride0*/ CHUNK * C_, /*pad: 64dw*/ 5, /*4dw*/ 3);
    // V chunk: 2D [128 c rows][32 m], global row stride N; pad 4 DWORDs every
    // 16 DWORDs (one 64B row) -> LDS row stride 40 halves.
    tdm_load2d((unsigned)(uintptr_t)&vbuf[bufi][0],
               Vc + (size_t)b * C_ * N_ + chunk * CHUNK,
               /*tile0*/ CHUNK, /*tile1*/ C_,
               /*tdim0*/ N_, /*tdim1*/ C_,
               /*stride0*/ N_, /*pad: 16dw*/ 3, /*4dw*/ 3);
  };

  if (wid == 0) { issue(0); issue(1); }
  const int iters = N_ / CHUNK;

  for (int it = 0; it < iters; ++it) {
    if (wid == 0) {
      if (it + 1 < iters) __builtin_amdgcn_s_wait_tensorcnt(2);
      else                __builtin_amdgcn_s_wait_tensorcnt(0);
    }
    __syncthreads();
    const _Float16* kb = kbuf[it & 1];
    const _Float16* vb = vbuf[it & 1];

    // S[n, m] for two 16-wide m subtiles; K-dim = c (4 x 32).
    v8f s0 = vz, s1 = vz;
    {
      const _Float16* kr0 = kb + l16 * KP + (hi ? 16 : 0);
      const _Float16* kr1 = kb + (16 + l16) * KP + (hi ? 16 : 0);
      #pragma unroll
      for (int kk = 0; kk < 4; ++kk) {
        s0 = wmma_f16(qf[kk], frag_16(kr0 + kk * 32), s0);
        s1 = wmma_f16(qf[kk], frag_16(kr1 + kk * 32), s1);
      }
    }

    // Online softmax: row r+8*hi lives in VGPR slot r, cols across 16 lanes.
    float scale[8];
    #pragma unroll
    for (int r = 0; r < 8; ++r) {
      float mx   = rmax16(fmaxf(s0[r], s1[r]));
      float mnew = fmaxf(m_i[r], mx);
      float sc   = __expf(m_i[r] - mnew);
      float p0   = __expf(s0[r] - mnew);
      float p1   = __expf(s1[r] - mnew);
      s0[r] = p0; s1[r] = p1;
      l_i[r]   = l_i[r] * sc + rsum16(p0 + p1);
      m_i[r]   = mnew;
      scale[r] = sc;
    }
    #pragma unroll
    for (int ct = 0; ct < 8; ++ct)
      #pragma unroll
      for (int r = 0; r < 8; ++r) o_acc[ct][r] *= scale[r];

    // Transpose P (D-layout -> A-layout) through per-wave LDS scratch.
    _Float16* pb = pbuf[wid];
    #pragma unroll
    for (int r = 0; r < 8; ++r) {
      pb[(r + 8 * hi) * PP + l16]      = (_Float16)s0[r];
      pb[(r + 8 * hi) * PP + 16 + l16] = (_Float16)s1[r];
    }
    asm volatile("s_wait_dscnt 0" ::: "memory");
    v16h pf;
    {
      const _Float16* pr = pb + l16 * PP + (hi ? 16 : 0);
      pf = frag_2x8(pr, pr + 8);
    }

    // O[n, c] += P[n, m] * V[c, m]  (B-fragment from vbuf[c][m]).
    #pragma unroll
    for (int ct = 0; ct < 8; ++ct) {
      const _Float16* vr = vb + (ct * 16 + l16) * VP + (hi ? 16 : 0);
      o_acc[ct] = wmma_f16(pf, frag_16(vr), o_acc[ct]);
    }
    __syncthreads();
    if (wid == 0 && it + 2 < iters) issue(it + 2);
  }

  // Normalize and store O token-major f16.
  #pragma unroll
  for (int ct = 0; ct < 8; ++ct) {
    #pragma unroll
    for (int r = 0; r < 8; ++r) {
      const int n = nrow0 + r + 8 * hi;
      Oh[((size_t)(b * N_ + n)) * C_ + ct * 16 + l16] =
          (_Float16)(o_acc[ct][r] / l_i[r]);
    }
  }
}

// ============================================================================
// Kernel 3: output projection  out[b,o,n] = sum_c wo[o,c]*O[n,c] + bo[o]  (fp32)
// ============================================================================
__global__ __launch_bounds__(256) void out_proj(
    const _Float16* __restrict__ Oh, const float* __restrict__ wo,
    const float* __restrict__ bo, float* __restrict__ out) {
  const int tid  = threadIdx.x;
  const int lane = tid & 31, wid = tid >> 5;
  const int l16  = lane & 15, hi = lane >> 4;
  const int b    = blockIdx.x >> 6;
  const int n0   = (blockIdx.x & 63) * 64;
  const int o_base = wid * 16;

  v16h af[4];
  {
    const int o = o_base + l16;
    const int base = hi ? 16 : 0;
    #pragma unroll
    for (int kk = 0; kk < 4; ++kk) {
      HF16 t;
      #pragma unroll
      for (int j = 0; j < 8; ++j) {
        t.h[j]     = (_Float16)wo[o * C_ + kk * 32 + base + j];
        t.h[8 + j] = (_Float16)wo[o * C_ + kk * 32 + base + 8 + j];
      }
      af[kk] = t.v;
    }
  }
  #pragma unroll
  for (int sub = 0; sub < 4; ++sub) {
    v8f acc = {0.f, 0.f, 0.f, 0.f, 0.f, 0.f, 0.f, 0.f};
    const _Float16* orow =
        Oh + ((size_t)(b * N_ + n0 + sub * 16 + l16)) * C_ + (hi ? 16 : 0);
    #pragma unroll
    for (int kk = 0; kk < 4; ++kk)
      acc = wmma_f16(af[kk], frag_16(orow + kk * 32), acc);
    const int n  = n0 + sub * 16 + l16;
    const int o0 = o_base + 8 * hi;
    #pragma unroll
    for (int rr = 0; rr < 8; ++rr)
      out[((size_t)(b * C_ + o0 + rr)) * N_ + n] = acc[rr] + bo[o0 + rr];
  }
}

extern "C" void kernel_launch(void* const* d_in, const int* in_sizes, int n_in,
                              void* d_out, int out_size, void* d_ws, size_t ws_size,
                              hipStream_t stream) {
  const float* x  = (const float*)d_in[0];
  const float* fw = (const float*)d_in[1];
  const float* wq = (const float*)d_in[2];
  const float* bq = (const float*)d_in[3];
  const float* wk = (const float*)d_in[4];
  const float* bk = (const float*)d_in[5];
  const float* wv = (const float*)d_in[6];
  const float* bv = (const float*)d_in[7];
  const float* wo = (const float*)d_in[8];
  const float* bo = (const float*)d_in[9];

  _Float16* Qh = (_Float16*)d_ws;                // [B][N][C] f16, pre-scaled
  _Float16* Kh = Qh + (size_t)B_ * N_ * C_;      // [B][N][C] f16
  _Float16* Vc = Kh + (size_t)B_ * N_ * C_;      // [B][C][N] f16
  _Float16* Oh = Vc + (size_t)B_ * N_ * C_;      // [B][N][C] f16

  proj_qkv<<<B_ * 64, 256, 0, stream>>>(x, fw, wq, bq, wk, bk, wv, bv, Qh, Kh, Vc);
  flash_attn<<<B_ * 32, 256, 0, stream>>>(Qh, Kh, Vc, Oh);
  out_proj<<<B_ * 64, 256, 0, stream>>>(Oh, wo, bo, (float*)d_out);

  (void)in_sizes; (void)n_in; (void)out_size; (void)ws_size;
}
